// Generator_16939351016125
// MI455X (gfx1250) — matrix operand order repaired and checked by reading, not compiled, not run on hardware
//
#include <hip/hip_runtime.h>
#include <hip/hip_bf16.h>

typedef __attribute__((ext_vector_type(16))) __bf16 v16bf;
typedef __attribute__((ext_vector_type(8)))  float  v8f;

union FragB {
    v16bf v;
    uint4 q[2];
};

__device__ __forceinline__ unsigned short f2bf(float f) {
    unsigned u = __float_as_uint(f);
    u += 0x7fffu + ((u >> 16) & 1u);   // round-to-nearest-even
    return (unsigned short)(u >> 16);
}

// ---------------------------------------------------------------------------
// Prep: build layer-1 input act0 as NHWC bf16 [128][2][2][544]
//   c in [0,512): x (NCHW f32), c in [512,517): landmark heatmap, else 0 pad.
// ---------------------------------------------------------------------------
__global__ void prep_input(const float* __restrict__ x, const int* __restrict__ lm,
                           unsigned short* __restrict__ act) {
    const int T = 128 * 2 * 2 * 544;
    int i = blockIdx.x * blockDim.x + threadIdx.x;
    if (i >= T) return;
    int n  = i / (2 * 2 * 544);
    int r  = i % (2 * 2 * 544);
    int yy = r / (2 * 544);
    int r2 = r % (2 * 544);
    int xx = r2 / 544;
    int c  = r2 % 544;
    float v;
    if (c < 512) {
        v = x[((n * 512 + c) * 2 + yy) * 2 + xx];
    } else if (c < 517) {
        int l  = c - 512;
        int lx = lm[(n * 5 + l) * 2 + 0] >> 6;   // int(v/128*2) for v in [0,128)
        int ly = lm[(n * 5 + l) * 2 + 1] >> 6;
        v = (lx == xx && ly == yy) ? 1.0f : 0.0f;
    } else {
        v = 0.0f;   // channel padding to multiple of 32
    }
    act[i] = f2bf(v);   // i already equals ((n*2+yy)*2+xx)*544 + c
}

// ---------------------------------------------------------------------------
// Pack weights (Cin,Cout,4,4) f32 -> bf16 A-fragments in WMMA lane order.
// Apack element index = ((t*MT + m)*KT + kk)*512 + lane*16 + j
// A layout (matches B fragment K order): K(lane,j) = j + 16*(lane>>4)
// ---------------------------------------------------------------------------
template <int CIN, int COUT, int CINP, int COUTP>
__global__ void pack_w(const float* __restrict__ w, unsigned short* __restrict__ ap) {
    constexpr int KT = CINP / 32, MT = COUTP / 16;
    constexpr int T = 16 * COUTP * CINP;
    int i = blockIdx.x * blockDim.x + threadIdx.x;
    if (i >= T) return;
    int j    = i & 15;
    int lane = (i >> 4) & 31;
    int rest = i >> 9;
    int kk   = rest % KT; rest /= KT;
    int m    = rest % MT;
    int t    = rest / MT;
    int co = m * 16 + (lane & 15);
    int K  = j + 16 * (lane >> 4);
    int ci = kk * 32 + K;
    float v = (ci < CIN && co < COUT)
                  ? w[(((size_t)ci * COUT + co) * 4 + (t >> 2)) * 4 + (t & 3)]
                  : 0.0f;
    ap[i] = f2bf(v);
}

// ---------------------------------------------------------------------------
// Implicit-GEMM transposed conv via V_WMMA_F32_16X16X32_BF16.
// grid = (COUTP/(16*COT), N*HIN*WIN/16, 4 parities); 1 wave32 per block.
// Parity (py,px): y-taps py==0 -> {(ky=1,dy=0),(ky=3,dy=-1)} else {(0,+1),(2,0)}.
// B fragment: lane<16 holds column N=lane, K=0..15; lane>=16 same column K=16..31
//   -> 16 consecutive bf16 in NHWC = two aligned b128 loads.
// ---------------------------------------------------------------------------
template <int CINP, int COUT, int COUTP, int HIN, int WIN, int COT, bool FINAL>
__global__ __launch_bounds__(32) void deconv_wmma(
    const unsigned short* __restrict__ act,    // NHWC bf16 [128][HIN][WIN][CINP]
    const unsigned short* __restrict__ apack,  // packed bf16 A fragments
    const float* __restrict__ bias,
    float* __restrict__ out)                   // NCHW f32 [128][COUT][2H][2W]
{
    constexpr int HOUT = 2 * HIN, WOUT = 2 * WIN;
    constexpr int KT = CINP / 32, MT = COUTP / 16;

    const int lane = threadIdx.x & 31;
    const int col  = lane & 15;
    const int hi   = lane >> 4;

    const int q   = blockIdx.y * 16 + col;          // pixel index (n,yi,xi)
    const int n   = q / (HIN * WIN);
    const int rem = q % (HIN * WIN);
    const int yi  = rem / WIN;
    const int xi  = rem % WIN;

    const int py = blockIdx.z >> 1, px = blockIdx.z & 1;
    const int dys[2] = { py ? 1 : 0, py ? 0 : -1 };
    const int kys[2] = { py ? 0 : 1, py ? 2 : 3 };
    const int dxs[2] = { px ? 1 : 0, px ? 0 : -1 };
    const int kxs[2] = { px ? 0 : 1, px ? 2 : 3 };

    v8f acc[COT];
#pragma unroll
    for (int s = 0; s < COT; ++s) acc[s] = v8f{0.f, 0.f, 0.f, 0.f, 0.f, 0.f, 0.f, 0.f};

    const int mTile0 = blockIdx.x * COT;

#pragma unroll
    for (int ti = 0; ti < 2; ++ti) {
        const int yy = yi + dys[ti];
#pragma unroll
        for (int tj = 0; tj < 2; ++tj) {
            const int xx  = xi + dxs[tj];
            const bool inb = (yy >= 0) && (yy < HIN) && (xx >= 0) && (xx < WIN);
            const int cy = inb ? yy : 0, cx = inb ? xx : 0;
            const unsigned short* bptr =
                act + (size_t)((n * HIN + cy) * WIN + cx) * CINP;
            const int t = kys[ti] * 4 + kxs[tj];
            const unsigned short* aptr =
                apack + ((size_t)(t * MT + mTile0)) * KT * 512 + lane * 16;

            for (int kk = 0; kk < KT; ++kk) {
                FragB bf;
                if (inb) {
                    const uint4* p = (const uint4*)(bptr + kk * 32 + hi * 16);
                    bf.q[0] = p[0];
                    bf.q[1] = p[1];
                } else {
                    bf.q[0] = make_uint4(0u, 0u, 0u, 0u);
                    bf.q[1] = make_uint4(0u, 0u, 0u, 0u);
                }
#pragma unroll
                for (int s = 0; s < COT; ++s) {
                    FragB af;
                    const uint4* pa = (const uint4*)(aptr + ((size_t)s * KT + kk) * 512);
                    af.q[0] = pa[0];
                    af.q[1] = pa[1];
                    acc[s] = __builtin_amdgcn_wmma_f32_16x16x32_bf16(
                        false, af.v, false, bf.v, (short)0, acc[s], false, false);
                }
            }
        }
    }

    // Epilogue: C/D layout -> lane<16: (M=r,N=lane); lane>=16: (M=r+8,N=lane-16)
    const int oy = 2 * yi + py, ox = 2 * xi + px;
#pragma unroll
    for (int s = 0; s < COT; ++s) {
#pragma unroll
        for (int r = 0; r < 8; ++r) {
            const int co = (mTile0 + s) * 16 + r + 8 * hi;
            if (FINAL) {
                if (co < COUT) {
                    float v = acc[s][r] + bias[co];
                    out[(((size_t)n * COUT + co) * HOUT + oy) * WOUT + ox] = tanhf(v);
                }
            } else {
                float v = acc[s][r] + bias[co];
                out[(((size_t)n * COUT + co) * HOUT + oy) * WOUT + ox] = v;
            }
        }
    }
}

// ---------------------------------------------------------------------------
// BatchNorm (training mode, biased var) statistics: one block per channel.
// ---------------------------------------------------------------------------
template <int COUT, int HO, int WO>
__global__ void bn_stats(const float* __restrict__ y, float* __restrict__ stats) {
    constexpr int M = 128 * HO * WO;
    __shared__ float ss[256], sq[256];
    const int c = blockIdx.x, tid = threadIdx.x;
    float s = 0.f, s2 = 0.f;
    for (int i = tid; i < M; i += 256) {
        int nn = i / (HO * WO);
        int p  = i % (HO * WO);
        float v = y[((size_t)nn * COUT + c) * (HO * WO) + p];
        s += v;
        s2 += v * v;
    }
    ss[tid] = s; sq[tid] = s2;
    __syncthreads();
    for (int st = 128; st > 0; st >>= 1) {
        if (tid < st) { ss[tid] += ss[tid + st]; sq[tid] += sq[tid + st]; }
        __syncthreads();
    }
    if (tid == 0) {
        float m   = ss[0] / (float)M;
        float var = sq[0] / (float)M - m * m;
        stats[2 * c]     = m;
        stats[2 * c + 1] = rsqrtf(var + 1e-5f);
    }
}

// ---------------------------------------------------------------------------
// BN apply + ReLU + f32->bf16 + NCHW->NHWC (next layer's GEMM input layout).
// ---------------------------------------------------------------------------
template <int COUT, int HO, int WO>
__global__ void bn_apply(const float* __restrict__ y, const float* __restrict__ stats,
                         const float* __restrict__ g, const float* __restrict__ be,
                         unsigned short* __restrict__ actn) {
    const int T = 128 * COUT * HO * WO;
    int i = blockIdx.x * blockDim.x + threadIdx.x;
    if (i >= T) return;
    int n  = i / (COUT * HO * WO);
    int r1 = i % (COUT * HO * WO);
    int c  = r1 / (HO * WO);
    int p  = r1 % (HO * WO);
    int yy = p / WO, xx = p % WO;
    float v = y[i];
    v = (v - stats[2 * c]) * stats[2 * c + 1] * g[c] + be[c];
    v = v > 0.f ? v : 0.f;
    actn[(((size_t)n * HO + yy) * WO + xx) * COUT + c] = f2bf(v);
}

// ---------------------------------------------------------------------------
extern "C" void kernel_launch(void* const* d_in, const int* in_sizes, int n_in,
                              void* d_out, int out_size, void* d_ws, size_t ws_size,
                              hipStream_t stream) {
    const float* x  = (const float*)d_in[0];
    const int*   lm = (const int*)d_in[1];
    const float *w1 = (const float*)d_in[2],  *b1 = (const float*)d_in[3],
                *g1 = (const float*)d_in[4],  *be1 = (const float*)d_in[5];
    const float *w2 = (const float*)d_in[6],  *b2 = (const float*)d_in[7],
                *g2 = (const float*)d_in[8],  *be2 = (const float*)d_in[9];
    const float *w3 = (const float*)d_in[10], *b3 = (const float*)d_in[11],
                *g3 = (const float*)d_in[12], *be3 = (const float*)d_in[13];
    const float *w4 = (const float*)d_in[14], *b4 = (const float*)d_in[15],
                *g4 = (const float*)d_in[16], *be4 = (const float*)d_in[17];
    const float *w5 = (const float*)d_in[18], *b5 = (const float*)d_in[19],
                *g5 = (const float*)d_in[20], *be5 = (const float*)d_in[21];
    const float *w6 = (const float*)d_in[22], *b6 = (const float*)d_in[23];
    float* out = (float*)d_out;

    char* ws = (char*)d_ws;
    // workspace layout (bytes)
    const size_t AP1 = (size_t)16 * 512 * 544 * 2;  // 8,912,896
    const size_t AP2 = (size_t)16 * 512 * 512 * 2;
    const size_t AP3 = (size_t)16 * 256 * 512 * 2;
    const size_t AP4 = (size_t)16 * 128 * 256 * 2;
    const size_t AP5 = (size_t)16 * 64 * 128 * 2;
    const size_t AP6 = (size_t)16 * 16 * 64 * 2;
    size_t off = 4096;                                   // [0,4096): BN stats
    float* stats = (float*)ws;
    unsigned short* ap1 = (unsigned short*)(ws + off); off += AP1;
    unsigned short* ap2 = (unsigned short*)(ws + off); off += AP2;
    unsigned short* ap3 = (unsigned short*)(ws + off); off += AP3;
    unsigned short* ap4 = (unsigned short*)(ws + off); off += AP4;
    unsigned short* ap5 = (unsigned short*)(ws + off); off += AP5;
    unsigned short* ap6 = (unsigned short*)(ws + off); off += AP6;
    const size_t ACT_B = (size_t)128 * 64 * 64 * 64 * 2; // 67,108,864
    unsigned short* actA = (unsigned short*)(ws + off); off += ACT_B;
    unsigned short* actB = (unsigned short*)(ws + off); off += ACT_B;
    float* ybuf = (float*)(ws + off);                    // 134,217,728 max (L5)

    // ---- weight packing (independent, issue up front) ----
    pack_w<517, 512, 544, 512><<<(16 * 512 * 544 + 255) / 256, 256, 0, stream>>>(w1, ap1);
    pack_w<512, 512, 512, 512><<<(16 * 512 * 512 + 255) / 256, 256, 0, stream>>>(w2, ap2);
    pack_w<512, 256, 512, 256><<<(16 * 256 * 512 + 255) / 256, 256, 0, stream>>>(w3, ap3);
    pack_w<256, 128, 256, 128><<<(16 * 128 * 256 + 255) / 256, 256, 0, stream>>>(w4, ap4);
    pack_w<128, 64, 128, 64><<<(16 * 64 * 128 + 255) / 256, 256, 0, stream>>>(w5, ap5);
    pack_w<64, 3, 64, 16><<<(16 * 16 * 64 + 255) / 256, 256, 0, stream>>>(w6, ap6);

    // ---- input prep: x + landmark heatmap -> NHWC bf16 padded to 544 ch ----
    prep_input<<<(128 * 2 * 2 * 544 + 255) / 256, 256, 0, stream>>>(x, lm, actA);

    // ---- layer 1: 544->512, 2x2 -> 4x4 ----
    deconv_wmma<544, 512, 512, 2, 2, 2, false>
        <<<dim3(16, 32, 4), 32, 0, stream>>>(actA, ap1, b1, ybuf);
    bn_stats<512, 4, 4><<<512, 256, 0, stream>>>(ybuf, stats);
    bn_apply<512, 4, 4><<<(128 * 512 * 16 + 255) / 256, 256, 0, stream>>>(ybuf, stats, g1, be1, actB);

    // ---- layer 2: 512->512, 4x4 -> 8x8 ----
    deconv_wmma<512, 512, 512, 4, 4, 2, false>
        <<<dim3(16, 128, 4), 32, 0, stream>>>(actB, ap2, b2, ybuf);
    bn_stats<512, 8, 8><<<512, 256, 0, stream>>>(ybuf, stats);
    bn_apply<512, 8, 8><<<(128 * 512 * 64 + 255) / 256, 256, 0, stream>>>(ybuf, stats, g2, be2, actA);

    // ---- layer 3: 512->256, 8x8 -> 16x16 ----
    deconv_wmma<512, 256, 256, 8, 8, 2, false>
        <<<dim3(8, 512, 4), 32, 0, stream>>>(actA, ap3, b3, ybuf);
    bn_stats<256, 16, 16><<<256, 256, 0, stream>>>(ybuf, stats);
    bn_apply<256, 16, 16><<<(128 * 256 * 256 + 255) / 256, 256, 0, stream>>>(ybuf, stats, g3, be3, actB);

    // ---- layer 4: 256->128, 16x16 -> 32x32 ----
    deconv_wmma<256, 128, 128, 16, 16, 2, false>
        <<<dim3(4, 2048, 4), 32, 0, stream>>>(actB, ap4, b4, ybuf);
    bn_stats<128, 32, 32><<<128, 256, 0, stream>>>(ybuf, stats);
    bn_apply<128, 32, 32><<<(128 * 128 * 1024 + 255) / 256, 256, 0, stream>>>(ybuf, stats, g4, be4, actA);

    // ---- layer 5: 128->64, 32x32 -> 64x64 ----
    deconv_wmma<128, 64, 64, 32, 32, 2, false>
        <<<dim3(2, 8192, 4), 32, 0, stream>>>(actA, ap5, b5, ybuf);
    bn_stats<64, 64, 64><<<64, 256, 0, stream>>>(ybuf, stats);
    bn_apply<64, 64, 64><<<(128 * 64 * 4096 + 255) / 256, 256, 0, stream>>>(ybuf, stats, g5, be5, actB);

    // ---- layer 6: 64->3, 64x64 -> 128x128, fused tanh -> d_out (f32) ----
    deconv_wmma<64, 3, 16, 64, 64, 1, true>
        <<<dim3(1, 32768, 4), 32, 0, stream>>>(actB, ap6, b6, out);
}
